// CrossAttention_C_18064632447176
// MI455X (gfx1250) — matrix operand order repaired and checked by reading, not compiled, not run on hardware
//
#include <hip/hip_runtime.h>

typedef __attribute__((ext_vector_type(16))) _Float16 v16h;
typedef __attribute__((ext_vector_type(8)))  _Float16 v8h;
typedef __attribute__((ext_vector_type(8)))  float    v8f;
typedef __attribute__((ext_vector_type(4)))  unsigned int v4u;
typedef __attribute__((ext_vector_type(4)))  int      v4i;
typedef __attribute__((ext_vector_type(8)))  int      v8i;

#define B_  8
#define C_  256
#define CR_ 64
#define N_  4096   // H*W

#define KSTRIDE 80   // halfs/row of Kt_lds: 128B data + 32B pad = 160B
#define VSTRIDE 40   // halfs/row of V_lds:   64B data + 16B pad =  80B
#define PSTRIDE 48   // halfs/row of P_lds (96B, 32B-aligned fragment reads)

#if defined(__has_builtin)
#if __has_builtin(__builtin_amdgcn_tensor_load_to_lds) && __has_builtin(__builtin_amdgcn_s_wait_tensorcnt)
#define HAVE_TDM 1
#if __has_include(<hip/amd_detail/amd_gfx1250_TDM.h>)
#define TDM_6ARG 1   // therock headers -> 6-arg builtin
#endif
#endif
#endif

// A-fragment (f16 16x32) K index for element e, lane half-group g (ISA 7.12.2)
__device__ __forceinline__ int a_kmap(int e, int g) {
    return (e < 8 ? e : e + 8) + 8 * g;
}

// LDS byte offset = low 32 bits of the flat address (ISA 10.2 aperture mapping)
__device__ __forceinline__ unsigned lds_off(const void* p) {
    return (unsigned)(unsigned long long)p;
}

// Fallback async DMA: per-lane 16B global -> LDS, tracked by ASYNCcnt
__device__ __forceinline__ void async_copy_b128(const _Float16* g, _Float16* l) {
    asm volatile("global_load_async_to_lds_b128 %0, %1, off"
                 :: "v"(lds_off(l)), "v"((unsigned long long)g)
                 : "memory");
}

#if defined(HAVE_TDM)
// One 2-D TDM descriptor: tile (tile0 x tile1) of a (dim0 x dim1) f16 tensor,
// row stride stride0 (elements), LDS destination padded per padi/pada codes.
__device__ __forceinline__ void tdm_load_2d(unsigned lds_addr, const void* gaddr,
                                            unsigned dim0, unsigned dim1,
                                            unsigned tile0, unsigned tile1,
                                            unsigned stride0,
                                            unsigned padi, unsigned pada)
{
    unsigned long long ga = (unsigned long long)gaddr;
    v4u g0 = { 1u,                                   // count=1, user descriptor
               lds_addr,                             // LDS byte address
               (unsigned)ga,                         // global_addr[31:0]
               (unsigned)((ga >> 32) & 0x01FFFFFFu)  // global_addr[56:32]
                 | 0x80000000u };                    // type=2 ("image")
    v8i g1 = { (int)((1u << 16)                      // data_size = 2 bytes
                   | (1u << 20)                      // pad_enable
                   | (padi << 22) | (pada << 25)),
               (int)((dim0 & 0xFFFFu) << 16),        // tensor_dim0[15:0] @ b63:48
               (int)(((dim0 >> 16) & 0xFFFFu)        // tensor_dim0[31:16]
                   | ((dim1 & 0xFFFFu) << 16)),      // tensor_dim1[15:0]
               (int)(((dim1 >> 16) & 0xFFFFu)        // tensor_dim1[31:16]
                   | (tile0 << 16)),                 // tile_dim0
               (int)(tile1 & 0xFFFFu),               // tile_dim1 (tile_dim2=0)
               (int)stride0,                         // tensor_dim0_stride[31:0]
               0, 0 };
    v4i gz = { 0, 0, 0, 0 };
#if defined(TDM_6ARG)
    v8i gz8 = { 0, 0, 0, 0, 0, 0, 0, 0 };
    __builtin_amdgcn_tensor_load_to_lds(g0, g1, gz, gz, gz8, 0);
#else
    __builtin_amdgcn_tensor_load_to_lds(g0, g1, gz, gz, 0);
#endif
}
#endif

// ---------------------------------------------------------------------------
// Projection GEMM: Out[r,n] = sum_c W[r,c] * X[b,c,n] (+bias), f32 in, f16 out.
// mode 0: Out stored [B][R][N]; mode 1: Out stored transposed [B][N][R].
// ---------------------------------------------------------------------------
__global__ void __launch_bounds__(32)
proj_wmma(const float* __restrict__ W,     // [R x C_]
          const float* __restrict__ X,     // [B][C_][N_]
          const float* __restrict__ bias,  // [R] or nullptr
          _Float16* __restrict__ Out,
          int R, int mode)
{
    const int lane = threadIdx.x & 31;
    const int col  = lane & 15;
    const int g    = lane >> 4;
    const int n0   = blockIdx.x * 16;
    const int r0   = blockIdx.y * 16;
    const int b    = blockIdx.z;

    const float* Xb = X + (size_t)b * C_ * N_;

    v8f acc = {};
    for (int c0 = 0; c0 < C_; c0 += 32) {
        v16h a, bb;
#pragma unroll
        for (int e = 0; e < 16; ++e) {
            a[e]  = (_Float16)W[(size_t)(r0 + col) * C_ + c0 + a_kmap(e, g)];
            bb[e] = (_Float16)Xb[(size_t)(c0 + a_kmap(e, g)) * N_ + n0 + col];
        }
        acc = __builtin_amdgcn_wmma_f32_16x16x32_f16(false, a, false, bb,
                                                     (short)0, acc, false, false);
    }

#pragma unroll
    for (int j = 0; j < 8; ++j) {
        int   r = r0 + j + 8 * g;
        float v = acc[j] + (bias ? bias[r] : 0.0f);
        if (mode == 0)
            Out[((size_t)b * R + r) * N_ + n0 + col] = (_Float16)v;
        else
            Out[((size_t)b * N_ + n0 + col) * R + r] = (_Float16)v;
    }
}

// ---------------------------------------------------------------------------
// Flash attention, 4 waves / block. Block double-buffers 32-key K/V tiles in
// LDS via the Tensor Data Mover (TENSORcnt) or per-lane async-to-LDS DMA
// (ASYNCcnt fallback). Per step/wave: 4 WMMAs S = Q^T K, online softmax,
// 16 WMMAs O += V P^T. Final: gamma*O/l + source.
// ---------------------------------------------------------------------------
__global__ void __launch_bounds__(128)
attn_wmma(const _Float16* __restrict__ Qt,   // [B][N][CR]
          const _Float16* __restrict__ Kt,   // [B][N][CR]  keys, transposed
          const _Float16* __restrict__ Vh,   // [B][C][N]
          const float*    __restrict__ src,  // [B][C][N]
          const float*    __restrict__ gamma1,
          float*          __restrict__ out)  // [B][C][N]
{
    __shared__ __align__(32) _Float16 Kt_lds[2][32 * KSTRIDE];  // [buf][m][q]
    __shared__ __align__(16) _Float16 V_lds [2][C_ * VSTRIDE];  // [buf][c][m]
    __shared__ __align__(32) _Float16 P_lds [4][16 * PSTRIDE];  // per-wave
    __shared__ float scale_lds[4][16];
    __shared__ float lsum_lds [4][16];

    const int tid  = threadIdx.x;
    const int wave = tid >> 5;
    const int lane = tid & 31;
    const int col  = lane & 15;
    const int g    = lane >> 4;
    const int nq   = blockIdx.x * 64 + wave * 16;
    const int b    = blockIdx.y;

    const _Float16* Qb = Qt + ((size_t)b * N_ + nq) * CR_;
    const _Float16* Kb = Kt + (size_t)b * N_ * CR_;
    const _Float16* Vb = Vh + (size_t)b * C_ * N_;

    const v8f vzero = {};

    // Q A-fragments: 16 queries x 64 channels = 2 x (16x32 f16)
    v16h aq[2];
#pragma unroll
    for (int t = 0; t < 2; ++t)
#pragma unroll
        for (int e = 0; e < 16; ++e)
            aq[t][e] = Qb[(size_t)col * CR_ + 32 * t + e + 16 * g];

    v8f acc[16];
#pragma unroll
    for (int ct = 0; ct < 16; ++ct) acc[ct] = vzero;

    float rm[8], rl[8];
#pragma unroll
    for (int j = 0; j < 8; ++j) { rm[j] = -3.0e38f; rl[j] = 0.0f; }

    // ---- stage one 32-key step: K tile (32x64) + V tile (256x32) ----------
    auto stage = [&](int buf, int m0) {
#if defined(HAVE_TDM)
        if (wave == 0) {
            // K: tile 64(q) x 32(m rows) of [N][CR] tensor, 32B pad / 128B row
            tdm_load_2d(lds_off(&Kt_lds[buf][0]), Kb + (size_t)m0 * CR_,
                        CR_, N_, CR_, 32, CR_, 4u, 7u);
            // V: tile 32(m) x 256(c rows) of [C][N] tensor, 16B pad / 64B row
            tdm_load_2d(lds_off(&V_lds[buf][0]), Vb + m0,
                        N_, C_, 32, C_, N_, 3u, 3u);
        }
#else
#pragma unroll
        for (int i = 0; i < 2; ++i) {          // K: 256 x 16B chunks
            int idx = tid + 128 * i;
            int row = idx >> 3, ch = idx & 7;
            async_copy_b128(Kb + (size_t)(m0 + row) * CR_ + ch * 8,
                            &Kt_lds[buf][row * KSTRIDE + ch * 8]);
        }
#pragma unroll
        for (int i = 0; i < 8; ++i) {          // V: 1024 x 16B chunks
            int idx = tid + 128 * i;
            int row = idx >> 2, ch = idx & 3;
            const _Float16* gp = Vb + (size_t)row * N_ + m0 + ch * 8;
            async_copy_b128(gp, &V_lds[buf][row * VSTRIDE + ch * 8]);
            if (m0 + 64 < N_) __builtin_prefetch(gp + 64, 0, 0);
        }
#endif
    };

    const int NSTEP = N_ / 32;
    stage(0, 0);

    for (int step = 0; step < NSTEP; ++step) {
        const int cb = step & 1;
        const int m0 = step * 32;
        if (step + 1 < NSTEP) {
            stage(cb ^ 1, m0 + 32);
#if defined(HAVE_TDM)
            __builtin_amdgcn_s_wait_tensorcnt((short)2);   // prev 2 TDM ops done
#else
            asm volatile("s_wait_asynccnt 10" ::: "memory");
#endif
        } else {
#if defined(HAVE_TDM)
            __builtin_amdgcn_s_wait_tensorcnt((short)0);
#else
            asm volatile("s_wait_asynccnt 0" ::: "memory");
#endif
        }
        __syncthreads();

        // ---- S = Q^T K : two 16x16 key half-tiles -------------------------
        v8f s[2];
        s[0] = vzero; s[1] = vzero;
#pragma unroll
        for (int half = 0; half < 2; ++half) {
#pragma unroll
            for (int t = 0; t < 2; ++t) {
                v16h bk = *(const v16h*)&Kt_lds[cb][(16 * half + col) * KSTRIDE + 32 * t + 16 * g];
                s[half] = __builtin_amdgcn_wmma_f32_16x16x32_f16(false, aq[t], false, bk,
                                                                 (short)0, s[half], false, false);
            }
        }

        // ---- online softmax; rows n = j + 8*g striped across 16 lanes -----
        float scj[8];
#pragma unroll
        for (int j = 0; j < 8; ++j) {
            float s0 = s[0][j], s1 = s[1][j];
            float mx = fmaxf(s0, s1);
#pragma unroll
            for (int off = 1; off < 16; off <<= 1)
                mx = fmaxf(mx, __shfl_xor(mx, off));
            float nm = fmaxf(rm[j], mx);
            float sc = __expf(rm[j] - nm);
            float p0 = __expf(s0 - nm);
            float p1 = __expf(s1 - nm);
            float ps = p0 + p1;
#pragma unroll
            for (int off = 1; off < 16; off <<= 1)
                ps += __shfl_xor(ps, off);
            rl[j] = rl[j] * sc + ps;
            rm[j] = nm;
            scj[j] = sc;
            int n = j + 8 * g;
            P_lds[wave][n * PSTRIDE + col]      = (_Float16)p0;
            P_lds[wave][n * PSTRIDE + 16 + col] = (_Float16)p1;
        }
        if (col == 0) {
#pragma unroll
            for (int j = 0; j < 8; ++j) scale_lds[wave][j + 8 * g] = scj[j];
        }
        // wave-local LDS RAW: in-order, no barrier needed
        float sn = scale_lds[wave][col];
        v8h plo = *(const v8h*)&P_lds[wave][col * PSTRIDE + 8 * g];
        v8h phi = *(const v8h*)&P_lds[wave][col * PSTRIDE + 16 + 8 * g];
        v16h bp = __builtin_shufflevector(plo, phi, 0, 1, 2, 3, 4, 5, 6, 7,
                                          8, 9, 10, 11, 12, 13, 14, 15);

        // ---- O += V P^T over 16 channel tiles -----------------------------
#pragma unroll
        for (int ct = 0; ct < 16; ++ct) {
#pragma unroll
            for (int j = 0; j < 8; ++j) acc[ct][j] *= sn;
            v8h lo = *(const v8h*)&V_lds[cb][(16 * ct + col) * VSTRIDE + 8 * g];
            v8h hi = *(const v8h*)&V_lds[cb][(16 * ct + col) * VSTRIDE + 16 + 8 * g];
            v16h av = __builtin_shufflevector(lo, hi, 0, 1, 2, 3, 4, 5, 6, 7,
                                              8, 9, 10, 11, 12, 13, 14, 15);
            acc[ct] = __builtin_amdgcn_wmma_f32_16x16x32_f16(false, av, false, bp,
                                                             (short)0, acc[ct], false, false);
        }
        __syncthreads();
    }

    if (col == 0) {
#pragma unroll
        for (int j = 0; j < 8; ++j) lsum_lds[wave][j + 8 * g] = rl[j];
    }
    const float inv = 1.0f / lsum_lds[wave][col];
    const float gm  = gamma1[0];

#pragma unroll
    for (int ct = 0; ct < 16; ++ct)
#pragma unroll
        for (int j = 0; j < 8; ++j) {
            int    c   = 16 * ct + j + 8 * g;
            size_t idx = ((size_t)b * C_ + c) * N_ + nq + col;
            out[idx] = gm * (acc[ct][j] * inv) + src[idx];
        }
}

// ---------------------------------------------------------------------------
extern "C" void kernel_launch(void* const* d_in, const int* in_sizes, int n_in,
                              void* d_out, int out_size, void* d_ws, size_t ws_size,
                              hipStream_t stream)
{
    const float* source   = (const float*)d_in[0];
    const float* guidance = (const float*)d_in[1];
    const float* Wq       = (const float*)d_in[2];
    const float* Wk       = (const float*)d_in[3];
    const float* Wv       = (const float*)d_in[4];
    const float* bv       = (const float*)d_in[5];
    const float* gamma    = (const float*)d_in[6];
    float*       out      = (float*)d_out;

    // Workspace: Qt (4MB) | Kt (4MB) | Vh (16MB), all f16
    _Float16* Qt = (_Float16*)d_ws;                 // [B][N][CR]
    _Float16* Kt = Qt + (size_t)B_ * N_ * CR_;      // [B][N][CR] (transposed keys)
    _Float16* Vh = Kt + (size_t)B_ * N_ * CR_;      // [B][C][N]

    proj_wmma<<<dim3(N_ / 16, CR_ / 16, B_), dim3(32), 0, stream>>>(Wq, source,   nullptr, Qt, CR_, 1);
    proj_wmma<<<dim3(N_ / 16, CR_ / 16, B_), dim3(32), 0, stream>>>(Wk, guidance, nullptr, Kt, CR_, 1);
    proj_wmma<<<dim3(N_ / 16, C_  / 16, B_), dim3(32), 0, stream>>>(Wv, guidance, bv,      Vh, C_,  0);

    attn_wmma<<<dim3(N_ / 64, B_), dim3(128), 0, stream>>>(Qt, Kt, Vh, source, gamma, out);
}
